// MoE_63084479643710
// MI455X (gfx1250) — compile-verified, matrix operand order checked by reading, not run on hardware
//
#include <hip/hip_runtime.h>
#include <cstdint>
#include <cmath>

// ---------------- problem constants ----------------
#define BS 2
#define SLEN 2048
#define DIM 2048
#define HIDDEN 1408
#define NEXP 8
#define TOPK 2
#define NTOK (BS*SLEN)          // 4096 tokens
#define NSLOT (NTOK*TOPK)       // 8192 routed rows (exact: every token routes twice)
#define MAXB 72                 // max 128-row blocks over all experts (<= 64 + 7)

typedef __attribute__((ext_vector_type(16))) __bf16 bf16x16;
typedef __attribute__((ext_vector_type(8)))  __bf16 bf16x8;
typedef __attribute__((ext_vector_type(8)))  float  f32x8;
typedef __attribute__((ext_vector_type(4)))  float  f32x4;
typedef __attribute__((ext_vector_type(2)))  float  f32x2;
typedef __attribute__((ext_vector_type(4)))  int    i32x4;

// ---------------- CDNA5 async global->LDS (guarded) ----------------
#if defined(__gfx1250__) && __has_builtin(__builtin_amdgcn_global_load_async_to_lds_b128)
#define HAS_ASYNC_LDS 1
#else
#define HAS_ASYNC_LDS 0
#endif

#if HAS_ASYNC_LDS
#define AS1 __attribute__((address_space(1)))
#define AS3 __attribute__((address_space(3)))
__device__ __forceinline__ void async_cp16(void* l, const void* g) {
  // Builtin signature (from the compiler diagnostic): param 1 is a global
  // (AS1) pointer to v4i32; C-style casts perform const-strip + addrspacecast
  // (same pattern HIP headers use for __builtin_amdgcn_global_atomic_*).
  __builtin_amdgcn_global_load_async_to_lds_b128(
      (AS1 i32x4*)g, (AS3 i32x4*)l, 0, 0);
}
#endif

__device__ __forceinline__ void wait_async_lds() {
#if HAS_ASYNC_LDS
#if __has_builtin(__builtin_amdgcn_s_wait_asynccnt)
  __builtin_amdgcn_s_wait_asynccnt(0);
#else
  asm volatile("s_wait_asynccnt 0" ::: "memory");
#endif
#endif
}

// ---------------- f32 -> bf16 weight conversion ----------------
__global__ __launch_bounds__(256) void cvt_bf16_k(const float* __restrict__ s,
                                                  __bf16* __restrict__ d, int n) {
  int i = (blockIdx.x * 256 + threadIdx.x) * 8;
  if (i >= n) return;
  f32x4 a = *(const f32x4*)&s[i];
  f32x4 b = *(const f32x4*)&s[i + 4];
  bf16x8 o;
#pragma unroll
  for (int q = 0; q < 4; ++q) { o[q] = (__bf16)a[q]; o[q + 4] = (__bf16)b[q]; }
  *(bf16x8*)&d[i] = o;
}

// ---------------- init counters ----------------
__global__ void init_k(int* cnt, int* rnk) {
  int i = threadIdx.x;
  if (i < NEXP) { cnt[i] = 0; rnk[i] = 0; }
}

// ---------------- router: sigmoid scores + biased top-2 ----------------
__global__ __launch_bounds__(256) void router_k(const float* __restrict__ x,
                                                const float* __restrict__ gw,
                                                const float* __restrict__ bias,
                                                int* __restrict__ sel,
                                                float* __restrict__ scs,
                                                int* __restrict__ cnt) {
  const int lane = threadIdx.x & 31;
  const int wv   = threadIdx.x >> 5;
  const int t    = blockIdx.x * 8 + wv;   // one wave32 per token
  const float* xp = x + (size_t)t * DIM;
  float acc[NEXP];
#pragma unroll
  for (int e = 0; e < NEXP; ++e) acc[e] = 0.f;
  for (int d = lane; d < DIM; d += 32) {
    float xv = xp[d];
#pragma unroll
    for (int e = 0; e < NEXP; ++e) acc[e] += xv * gw[e * DIM + d];
  }
#pragma unroll
  for (int e = 0; e < NEXP; ++e)
#pragma unroll
    for (int off = 16; off; off >>= 1) acc[e] += __shfl_xor(acc[e], off, 32);
  if (lane == 0) {
    float s[NEXP], v[NEXP];
#pragma unroll
    for (int e = 0; e < NEXP; ++e) {
      s[e] = 1.f / (1.f + __expf(-acc[e]));
      v[e] = s[e] + bias[e];
    }
    int e1 = 0;
    for (int e = 1; e < NEXP; ++e) if (v[e] > v[e1]) e1 = e;   // lowest index wins ties
    int e2 = (e1 == 0) ? 1 : 0;
    for (int e = 0; e < NEXP; ++e) if (e != e1 && v[e] > v[e2]) e2 = e;
    sel[t * 2 + 0] = e1;  scs[t * 2 + 0] = s[e1];  // ROUTE_SCALE == 1.0
    sel[t * 2 + 1] = e2;  scs[t * 2 + 1] = s[e2];
    atomicAdd(&cnt[e1], 1);
    atomicAdd(&cnt[e2], 1);
  }
}

// ---------------- schedule: bases + per-block descriptors ----------------
__global__ void sched_k(const int* __restrict__ cnt, int* __restrict__ base,
                        int* __restrict__ sched) {
  int b = 0, acc = 0;
  for (int e = 0; e < NEXP; ++e) {
    base[e] = acc;
    int c = cnt[e];
    for (int r = 0; r < c; r += 128) {
      sched[b * 3 + 0] = e;
      sched[b * 3 + 1] = acc + r;
      sched[b * 3 + 2] = (c - r < 128) ? (c - r) : 128;
      ++b;
    }
    acc += c;
  }
  for (; b < MAXB; ++b) { sched[b*3] = 0; sched[b*3+1] = 0; sched[b*3+2] = 0; }
}

// ---------------- per-slot destination row ----------------
__global__ void pos_k(const int* __restrict__ sel, const int* __restrict__ base,
                      int* __restrict__ rnk, int* __restrict__ pos) {
  int s = blockIdx.x * 256 + threadIdx.x;
  if (s < NSLOT) {
    int e = sel[s];
    pos[s] = base[e] + atomicAdd(&rnk[e], 1);
  }
}

// ---------------- dispatch: routed_in[row] = bf16(x[t] * score) ----------------
__global__ __launch_bounds__(256) void dispatch_k(const float* __restrict__ x,
                                                  const float* __restrict__ scs,
                                                  const int* __restrict__ pos,
                                                  __bf16* __restrict__ Xr) {
  int s = blockIdx.x;
  int t = s >> 1;
  float sc = scs[s];
  const float* xp = x + (size_t)t * DIM;
  __bf16* dp = Xr + (size_t)pos[s] * DIM;
  for (int i = threadIdx.x * 8; i < DIM; i += 256 * 8) {
    f32x4 a = *(const f32x4*)&xp[i];
    f32x4 b = *(const f32x4*)&xp[i + 4];
    bf16x8 o;
#pragma unroll
    for (int q = 0; q < 4; ++q) { o[q] = (__bf16)(a[q] * sc); o[q+4] = (__bf16)(b[q] * sc); }
    *(bf16x8*)&dp[i] = o;
  }
}

// -------- bf16 fragment load from LDS (16-bit A/B layout, ISA 7.12.2) --------
__device__ __forceinline__ bf16x16 frag16(const __bf16* p) {
  bf16x8 lo = *(const bf16x8*)p;
  bf16x8 hi = *(const bf16x8*)(p + 16);
  bf16x16 f;
#pragma unroll
  for (int q = 0; q < 8; ++q) { f[q] = lo[q]; f[q + 8] = hi[q]; }
  return f;
}

// ---------------- routed GEMM1: h = silu(X W1^T) * (X W3^T), bf16 WMMA ----------------
// grid.x = HIDDEN/128 = 11 N-tiles, grid.y = MAXB row-blocks. Double-buffered LDS.
__global__ __launch_bounds__(256) void gemm1_routed(const __bf16* __restrict__ Xr,
                                                    const __bf16* __restrict__ W1,
                                                    const __bf16* __restrict__ W3,
                                                    __bf16* __restrict__ H,
                                                    const int* __restrict__ sched) {
  __shared__ __align__(16) __bf16 lA [2][128 * 40];
  __shared__ __align__(16) __bf16 lB1[2][128 * 40];
  __shared__ __align__(16) __bf16 lB3[2][128 * 40];
  const int b    = blockIdx.y;
  const int e    = sched[b * 3 + 0];
  const int row0 = sched[b * 3 + 1];
  const int rows = sched[b * 3 + 2];
  if (rows == 0) return;
  const int n0 = blockIdx.x * 128;
  const int tid = threadIdx.x, lane = tid & 31, wv = tid >> 5;
  const int wM = wv >> 2, wN = wv & 3;
  const __bf16* w1e = W1 + (size_t)e * HIDDEN * DIM;
  const __bf16* w3e = W3 + (size_t)e * HIDDEN * DIM;

  // staging coords: 512 16B chunks, 2 per thread
  const int r0c = tid >> 2,          cc0 = (tid & 3) * 8;
  const int r1c = (tid + 256) >> 2,  cc1 = ((tid + 256) & 3) * 8;
  int gr0 = row0 + r0c; if (gr0 > NSLOT - 1) gr0 = NSLOT - 1;
  int gr1 = row0 + r1c; if (gr1 > NSLOT - 1) gr1 = NSLOT - 1;

#if !HAS_ASYNC_LDS
  i32x4 st[6];
#endif
  auto fetch = [&](int p, int kt) {
#if HAS_ASYNC_LDS
    async_cp16(&lA [p][r0c*40+cc0], &Xr [(size_t)gr0*DIM + kt + cc0]);
    async_cp16(&lA [p][r1c*40+cc1], &Xr [(size_t)gr1*DIM + kt + cc1]);
    async_cp16(&lB1[p][r0c*40+cc0], &w1e[(size_t)(n0+r0c)*DIM + kt + cc0]);
    async_cp16(&lB1[p][r1c*40+cc1], &w1e[(size_t)(n0+r1c)*DIM + kt + cc1]);
    async_cp16(&lB3[p][r0c*40+cc0], &w3e[(size_t)(n0+r0c)*DIM + kt + cc0]);
    async_cp16(&lB3[p][r1c*40+cc1], &w3e[(size_t)(n0+r1c)*DIM + kt + cc1]);
#else
    (void)p;
    st[0] = *(const i32x4*)&Xr [(size_t)gr0*DIM + kt + cc0];
    st[1] = *(const i32x4*)&Xr [(size_t)gr1*DIM + kt + cc1];
    st[2] = *(const i32x4*)&w1e[(size_t)(n0+r0c)*DIM + kt + cc0];
    st[3] = *(const i32x4*)&w1e[(size_t)(n0+r1c)*DIM + kt + cc1];
    st[4] = *(const i32x4*)&w3e[(size_t)(n0+r0c)*DIM + kt + cc0];
    st[5] = *(const i32x4*)&w3e[(size_t)(n0+r1c)*DIM + kt + cc1];
#endif
  };
  auto commit = [&](int p) {
#if !HAS_ASYNC_LDS
    *(i32x4*)&lA [p][r0c*40+cc0] = st[0];
    *(i32x4*)&lA [p][r1c*40+cc1] = st[1];
    *(i32x4*)&lB1[p][r0c*40+cc0] = st[2];
    *(i32x4*)&lB1[p][r1c*40+cc1] = st[3];
    *(i32x4*)&lB3[p][r0c*40+cc0] = st[4];
    *(i32x4*)&lB3[p][r1c*40+cc1] = st[5];
#else
    (void)p;
#endif
  };

  f32x8 acc1[4][2] = {}; f32x8 acc3[4][2] = {};
  const int rsel = lane & 15, kb = (lane >> 4) * 8;

  fetch(0, 0); commit(0);
  wait_async_lds();
  __syncthreads();
  int p = 0;
  for (int kt = 0; kt < DIM; kt += 32) {
    const bool more = (kt + 32) < DIM;
    if (more) fetch(p ^ 1, kt + 32);

    bf16x16 af[4], b1f[2], b3f[2];
#pragma unroll
    for (int i = 0; i < 4; ++i) af[i] = frag16(&lA[p][(wM * 64 + i * 16 + rsel) * 40 + kb]);
#pragma unroll
    for (int j = 0; j < 2; ++j) {
      b1f[j] = frag16(&lB1[p][(wN * 32 + j * 16 + rsel) * 40 + kb]);
      b3f[j] = frag16(&lB3[p][(wN * 32 + j * 16 + rsel) * 40 + kb]);
    }
#pragma unroll
    for (int i = 0; i < 4; ++i)
#pragma unroll
      for (int j = 0; j < 2; ++j) {
        acc1[i][j] = __builtin_amdgcn_wmma_f32_16x16x32_bf16(false, af[i], false, b1f[j],
                                                             (short)0, acc1[i][j], false, false);
        acc3[i][j] = __builtin_amdgcn_wmma_f32_16x16x32_bf16(false, af[i], false, b3f[j],
                                                             (short)0, acc3[i][j], false, false);
      }
    if (more) commit(p ^ 1);
    wait_async_lds();
    __syncthreads();
    p ^= 1;
  }
  const int nloc = lane & 15, mAdd = (lane >> 4) * 8;
#pragma unroll
  for (int i = 0; i < 4; ++i)
#pragma unroll
    for (int j = 0; j < 2; ++j)
#pragma unroll
      for (int q = 0; q < 8; ++q) {
        int ml = wM * 64 + i * 16 + mAdd + q;
        if (ml < rows) {
          float a = acc1[i][j][q], c3 = acc3[i][j][q];
          float s = (a / (1.f + __expf(-a))) * c3;  // silu(a) * c3
          H[(size_t)(row0 + ml) * HIDDEN + n0 + wN * 32 + j * 16 + nloc] = (__bf16)s;
        }
      }
}

// ---------------- routed GEMM2: routed_out = h W2^T, bf16 WMMA -> f32 ----------------
// grid.x = DIM/128 = 16, grid.y = MAXB. Double-buffered LDS.
__global__ __launch_bounds__(256) void gemm2_routed(const __bf16* __restrict__ H,
                                                    const __bf16* __restrict__ W2,
                                                    float* __restrict__ RO,
                                                    const int* __restrict__ sched) {
  __shared__ __align__(16) __bf16 lA[2][128 * 40];
  __shared__ __align__(16) __bf16 lB[2][128 * 40];
  const int b    = blockIdx.y;
  const int e    = sched[b * 3 + 0];
  const int row0 = sched[b * 3 + 1];
  const int rows = sched[b * 3 + 2];
  if (rows == 0) return;
  const int n0 = blockIdx.x * 128;
  const int tid = threadIdx.x, lane = tid & 31, wv = tid >> 5;
  const int wM = wv >> 2, wN = wv & 3;
  const __bf16* w2e = W2 + (size_t)e * DIM * HIDDEN;

  const int r0c = tid >> 2,          cc0 = (tid & 3) * 8;
  const int r1c = (tid + 256) >> 2,  cc1 = ((tid + 256) & 3) * 8;
  int gr0 = row0 + r0c; if (gr0 > NSLOT - 1) gr0 = NSLOT - 1;
  int gr1 = row0 + r1c; if (gr1 > NSLOT - 1) gr1 = NSLOT - 1;

#if !HAS_ASYNC_LDS
  i32x4 st[4];
#endif
  auto fetch = [&](int p, int kt) {
#if HAS_ASYNC_LDS
    async_cp16(&lA[p][r0c*40+cc0], &H  [(size_t)gr0*HIDDEN + kt + cc0]);
    async_cp16(&lA[p][r1c*40+cc1], &H  [(size_t)gr1*HIDDEN + kt + cc1]);
    async_cp16(&lB[p][r0c*40+cc0], &w2e[(size_t)(n0+r0c)*HIDDEN + kt + cc0]);
    async_cp16(&lB[p][r1c*40+cc1], &w2e[(size_t)(n0+r1c)*HIDDEN + kt + cc1]);
#else
    (void)p;
    st[0] = *(const i32x4*)&H  [(size_t)gr0*HIDDEN + kt + cc0];
    st[1] = *(const i32x4*)&H  [(size_t)gr1*HIDDEN + kt + cc1];
    st[2] = *(const i32x4*)&w2e[(size_t)(n0+r0c)*HIDDEN + kt + cc0];
    st[3] = *(const i32x4*)&w2e[(size_t)(n0+r1c)*HIDDEN + kt + cc1];
#endif
  };
  auto commit = [&](int p) {
#if !HAS_ASYNC_LDS
    *(i32x4*)&lA[p][r0c*40+cc0] = st[0];
    *(i32x4*)&lA[p][r1c*40+cc1] = st[1];
    *(i32x4*)&lB[p][r0c*40+cc0] = st[2];
    *(i32x4*)&lB[p][r1c*40+cc1] = st[3];
#else
    (void)p;
#endif
  };

  f32x8 acc[4][2] = {};
  const int rsel = lane & 15, kb = (lane >> 4) * 8;

  fetch(0, 0); commit(0);
  wait_async_lds();
  __syncthreads();
  int p = 0;
  for (int kt = 0; kt < HIDDEN; kt += 32) {
    const bool more = (kt + 32) < HIDDEN;
    if (more) fetch(p ^ 1, kt + 32);

    bf16x16 af[4], bf[2];
#pragma unroll
    for (int i = 0; i < 4; ++i) af[i] = frag16(&lA[p][(wM * 64 + i * 16 + rsel) * 40 + kb]);
#pragma unroll
    for (int j = 0; j < 2; ++j) bf[j] = frag16(&lB[p][(wN * 32 + j * 16 + rsel) * 40 + kb]);
#pragma unroll
    for (int i = 0; i < 4; ++i)
#pragma unroll
      for (int j = 0; j < 2; ++j)
        acc[i][j] = __builtin_amdgcn_wmma_f32_16x16x32_bf16(false, af[i], false, bf[j],
                                                            (short)0, acc[i][j], false, false);
    if (more) commit(p ^ 1);
    wait_async_lds();
    __syncthreads();
    p ^= 1;
  }
  const int nloc = lane & 15, mAdd = (lane >> 4) * 8;
#pragma unroll
  for (int i = 0; i < 4; ++i)
#pragma unroll
    for (int j = 0; j < 2; ++j)
#pragma unroll
      for (int q = 0; q < 8; ++q) {
        int ml = wM * 64 + i * 16 + mAdd + q;
        if (ml < rows)
          RO[(size_t)(row0 + ml) * DIM + n0 + wN * 32 + j * 16 + nloc] = acc[i][j][q];
      }
}

// ---------------- shared expert GEMM1 (f32 WMMA 16x16x4) ----------------
// grid.x = HIDDEN/128 = 11, grid.y = NTOK/128 = 32. Double-buffered LDS.
__global__ __launch_bounds__(256) void sgemm1_k(const float* __restrict__ X,
                                                const float* __restrict__ W1,
                                                const float* __restrict__ W3,
                                                float* __restrict__ Hs) {
  __shared__ __align__(16) float lA [2][128 * 20];
  __shared__ __align__(16) float lB1[2][128 * 20];
  __shared__ __align__(16) float lB3[2][128 * 20];
  const int n0 = blockIdx.x * 128;
  const int m0 = blockIdx.y * 128;
  const int tid = threadIdx.x, lane = tid & 31, wv = tid >> 5;
  const int wM = wv >> 2, wN = wv & 3;

  const int r0c = tid >> 2,          cc0 = (tid & 3) * 4;
  const int r1c = (tid + 256) >> 2,  cc1 = ((tid + 256) & 3) * 4;

#if !HAS_ASYNC_LDS
  i32x4 st[6];
#endif
  auto fetch = [&](int p, int kt) {
#if HAS_ASYNC_LDS
    async_cp16(&lA [p][r0c*20+cc0], &X [(size_t)(m0+r0c)*DIM + kt + cc0]);
    async_cp16(&lA [p][r1c*20+cc1], &X [(size_t)(m0+r1c)*DIM + kt + cc1]);
    async_cp16(&lB1[p][r0c*20+cc0], &W1[(size_t)(n0+r0c)*DIM + kt + cc0]);
    async_cp16(&lB1[p][r1c*20+cc1], &W1[(size_t)(n0+r1c)*DIM + kt + cc1]);
    async_cp16(&lB3[p][r0c*20+cc0], &W3[(size_t)(n0+r0c)*DIM + kt + cc0]);
    async_cp16(&lB3[p][r1c*20+cc1], &W3[(size_t)(n0+r1c)*DIM + kt + cc1]);
#else
    (void)p;
    st[0] = *(const i32x4*)&X [(size_t)(m0+r0c)*DIM + kt + cc0];
    st[1] = *(const i32x4*)&X [(size_t)(m0+r1c)*DIM + kt + cc1];
    st[2] = *(const i32x4*)&W1[(size_t)(n0+r0c)*DIM + kt + cc0];
    st[3] = *(const i32x4*)&W1[(size_t)(n0+r1c)*DIM + kt + cc1];
    st[4] = *(const i32x4*)&W3[(size_t)(n0+r0c)*DIM + kt + cc0];
    st[5] = *(const i32x4*)&W3[(size_t)(n0+r1c)*DIM + kt + cc1];
#endif
  };
  auto commit = [&](int p) {
#if !HAS_ASYNC_LDS
    *(i32x4*)&lA [p][r0c*20+cc0] = st[0];
    *(i32x4*)&lA [p][r1c*20+cc1] = st[1];
    *(i32x4*)&lB1[p][r0c*20+cc0] = st[2];
    *(i32x4*)&lB1[p][r1c*20+cc1] = st[3];
    *(i32x4*)&lB3[p][r0c*20+cc0] = st[4];
    *(i32x4*)&lB3[p][r1c*20+cc1] = st[5];
#else
    (void)p;
#endif
  };

  f32x8 acc1[4][2] = {}; f32x8 acc3[4][2] = {};
  const int rsel = lane & 15, kadd = (lane >> 4) * 2;

  fetch(0, 0); commit(0);
  wait_async_lds();
  __syncthreads();
  int p = 0;
  for (int kt = 0; kt < DIM; kt += 16) {
    const bool more = (kt + 16) < DIM;
    if (more) fetch(p ^ 1, kt + 16);
#pragma unroll
    for (int kk = 0; kk < 16; kk += 4) {
      f32x2 af[4], b1f[2], b3f[2];
#pragma unroll
      for (int i = 0; i < 4; ++i) af[i] = *(const f32x2*)&lA[p][(wM*64 + i*16 + rsel)*20 + kk + kadd];
#pragma unroll
      for (int j = 0; j < 2; ++j) {
        b1f[j] = *(const f32x2*)&lB1[p][(wN*32 + j*16 + rsel)*20 + kk + kadd];
        b3f[j] = *(const f32x2*)&lB3[p][(wN*32 + j*16 + rsel)*20 + kk + kadd];
      }
#pragma unroll
      for (int i = 0; i < 4; ++i)
#pragma unroll
        for (int j = 0; j < 2; ++j) {
          acc1[i][j] = __builtin_amdgcn_wmma_f32_16x16x4_f32(false, af[i], false, b1f[j],
                                                             (short)0, acc1[i][j], false, false);
          acc3[i][j] = __builtin_amdgcn_wmma_f32_16x16x4_f32(false, af[i], false, b3f[j],
                                                             (short)0, acc3[i][j], false, false);
        }
    }
    if (more) commit(p ^ 1);
    wait_async_lds();
    __syncthreads();
    p ^= 1;
  }
  const int nloc = lane & 15, mAdd = (lane >> 4) * 8;
#pragma unroll
  for (int i = 0; i < 4; ++i)
#pragma unroll
    for (int j = 0; j < 2; ++j)
#pragma unroll
      for (int q = 0; q < 8; ++q) {
        float a = acc1[i][j][q], c3 = acc3[i][j][q];
        Hs[(size_t)(m0 + wM*64 + i*16 + mAdd + q) * HIDDEN + n0 + wN*32 + j*16 + nloc] =
            (a / (1.f + __expf(-a))) * c3;
      }
}

// ---------------- shared expert GEMM2 (f32 WMMA) -> writes d_out ----------------
// grid.x = DIM/128 = 16, grid.y = NTOK/128 = 32. Double-buffered LDS.
__global__ __launch_bounds__(256) void sgemm2_k(const float* __restrict__ Hs,
                                                const float* __restrict__ W2,
                                                float* __restrict__ Out) {
  __shared__ __align__(16) float lA[2][128 * 20];
  __shared__ __align__(16) float lB[2][128 * 20];
  const int n0 = blockIdx.x * 128;
  const int m0 = blockIdx.y * 128;
  const int tid = threadIdx.x, lane = tid & 31, wv = tid >> 5;
  const int wM = wv >> 2, wN = wv & 3;

  const int r0c = tid >> 2,          cc0 = (tid & 3) * 4;
  const int r1c = (tid + 256) >> 2,  cc1 = ((tid + 256) & 3) * 4;

#if !HAS_ASYNC_LDS
  i32x4 st[4];
#endif
  auto fetch = [&](int p, int kt) {
#if HAS_ASYNC_LDS
    async_cp16(&lA[p][r0c*20+cc0], &Hs[(size_t)(m0+r0c)*HIDDEN + kt + cc0]);
    async_cp16(&lA[p][r1c*20+cc1], &Hs[(size_t)(m0+r1c)*HIDDEN + kt + cc1]);
    async_cp16(&lB[p][r0c*20+cc0], &W2[(size_t)(n0+r0c)*HIDDEN + kt + cc0]);
    async_cp16(&lB[p][r1c*20+cc1], &W2[(size_t)(n0+r1c)*HIDDEN + kt + cc1]);
#else
    (void)p;
    st[0] = *(const i32x4*)&Hs[(size_t)(m0+r0c)*HIDDEN + kt + cc0];
    st[1] = *(const i32x4*)&Hs[(size_t)(m0+r1c)*HIDDEN + kt + cc1];
    st[2] = *(const i32x4*)&W2[(size_t)(n0+r0c)*HIDDEN + kt + cc0];
    st[3] = *(const i32x4*)&W2[(size_t)(n0+r1c)*HIDDEN + kt + cc1];
#endif
  };
  auto commit = [&](int p) {
#if !HAS_ASYNC_LDS
    *(i32x4*)&lA[p][r0c*20+cc0] = st[0];
    *(i32x4*)&lA[p][r1c*20+cc1] = st[1];
    *(i32x4*)&lB[p][r0c*20+cc0] = st[2];
    *(i32x4*)&lB[p][r1c*20+cc1] = st[3];
#else
    (void)p;
#endif
  };

  f32x8 acc[4][2] = {};
  const int rsel = lane & 15, kadd = (lane >> 4) * 2;

  fetch(0, 0); commit(0);
  wait_async_lds();
  __syncthreads();
  int p = 0;
  for (int kt = 0; kt < HIDDEN; kt += 16) {
    const bool more = (kt + 16) < HIDDEN;
    if (more) fetch(p ^ 1, kt + 16);
#pragma unroll
    for (int kk = 0; kk < 16; kk += 4) {
      f32x2 af[4], bf[2];
#pragma unroll
      for (int i = 0; i < 4; ++i) af[i] = *(const f32x2*)&lA[p][(wM*64 + i*16 + rsel)*20 + kk + kadd];
#pragma unroll
      for (int j = 0; j < 2; ++j) bf[j] = *(const f32x2*)&lB[p][(wN*32 + j*16 + rsel)*20 + kk + kadd];
#pragma unroll
      for (int i = 0; i < 4; ++i)
#pragma unroll
        for (int j = 0; j < 2; ++j)
          acc[i][j] = __builtin_amdgcn_wmma_f32_16x16x4_f32(false, af[i], false, bf[j],
                                                            (short)0, acc[i][j], false, false);
    }
    if (more) commit(p ^ 1);
    wait_async_lds();
    __syncthreads();
    p ^= 1;
  }
  const int nloc = lane & 15, mAdd = (lane >> 4) * 8;
#pragma unroll
  for (int i = 0; i < 4; ++i)
#pragma unroll
    for (int j = 0; j < 2; ++j)
#pragma unroll
      for (int q = 0; q < 8; ++q)
        Out[(size_t)(m0 + wM*64 + i*16 + mAdd + q) * DIM + n0 + wN*32 + j*16 + nloc] =
            acc[i][j][q];
}

// ---------------- combine: out[t] += routed_out[pos0] + routed_out[pos1] ----------------
__global__ __launch_bounds__(256) void combine_k(const float* __restrict__ RO,
                                                 const int* __restrict__ pos,
                                                 float* __restrict__ Out) {
  int t = blockIdx.x;
  const float* r0 = RO + (size_t)pos[t * 2 + 0] * DIM;
  const float* r1 = RO + (size_t)pos[t * 2 + 1] * DIM;
  float* op = Out + (size_t)t * DIM;
  for (int i = threadIdx.x * 4; i < DIM; i += 256 * 4) {
    f32x4 v = *(f32x4*)&op[i];
    f32x4 a = *(const f32x4*)&r0[i];
    f32x4 b = *(const f32x4*)&r1[i];
    v = v + a + b;
    *(f32x4*)&op[i] = v;
  }
}

// ---------------- host launcher ----------------
extern "C" void kernel_launch(void* const* d_in, const int* in_sizes, int n_in,
                              void* d_out, int out_size, void* d_ws, size_t ws_size,
                              hipStream_t stream) {
  const float* x     = (const float*)d_in[0];
  const float* gatew = (const float*)d_in[1];
  const float* w1    = (const float*)d_in[2];
  const float* w2    = (const float*)d_in[3];
  const float* w3    = (const float*)d_in[4];
  const float* sw1   = (const float*)d_in[5];
  const float* sw2   = (const float*)d_in[6];
  const float* sw3   = (const float*)d_in[7];
  const float* ebias = (const float*)d_in[8];
  float* out = (float*)d_out;

  // workspace carve-up (256B aligned)
  char* w = (char*)d_ws;
  size_t off = 0;
  auto take = [&](size_t bytes) { char* p = w + off; off = (off + bytes + 255) & ~(size_t)255; return p; };
  const size_t WBYTES = (size_t)NEXP * HIDDEN * DIM * 2;   // 46,137,344
  __bf16* w1bf = (__bf16*)take(WBYTES);
  __bf16* w3bf = (__bf16*)take(WBYTES);
  __bf16* w2bf = (__bf16*)take(WBYTES);
  __bf16* Xr   = (__bf16*)take((size_t)NSLOT * DIM * 2);
  __bf16* Hb   = (__bf16*)take((size_t)NSLOT * HIDDEN * 2);
  float*  RO   = (float*)take((size_t)NSLOT * DIM * 4);
  float*  Hs   = (float*)take((size_t)NTOK * HIDDEN * 4);
  int* sel  = (int*)take(NSLOT * 4);
  float* scs = (float*)take(NSLOT * 4);
  int* pos  = (int*)take(NSLOT * 4);
  int* cnt  = (int*)take(256);
  int* rnk  = (int*)take(256);
  int* base = (int*)take(256);
  int* sched= (int*)take(MAXB * 3 * 4);
  (void)ws_size; (void)in_sizes; (void)n_in; (void)out_size;

  const int nw = NEXP * HIDDEN * DIM;                     // 23,068,672 (divisible by 2048)
  cvt_bf16_k<<<nw / 2048, 256, 0, stream>>>(w1, w1bf, nw);
  cvt_bf16_k<<<nw / 2048, 256, 0, stream>>>(w3, w3bf, nw);
  cvt_bf16_k<<<nw / 2048, 256, 0, stream>>>(w2, w2bf, nw);

  init_k<<<1, 64, 0, stream>>>(cnt, rnk);
  router_k<<<NTOK / 8, 256, 0, stream>>>(x, gatew, ebias, sel, scs, cnt);
  sched_k<<<1, 1, 0, stream>>>(cnt, base, sched);
  pos_k<<<NSLOT / 256, 256, 0, stream>>>(sel, base, rnk, pos);
  dispatch_k<<<NSLOT, 256, 0, stream>>>(x, scs, pos, Xr);

  gemm1_routed<<<dim3(HIDDEN / 128, MAXB), 256, 0, stream>>>(Xr, w1bf, w3bf, Hb, sched);
  gemm2_routed<<<dim3(DIM / 128, MAXB), 256, 0, stream>>>(Hb, w2bf, RO, sched);

  sgemm1_k<<<dim3(HIDDEN / 128, NTOK / 128), 256, 0, stream>>>(x, sw1, sw3, Hs);
  sgemm2_k<<<dim3(DIM / 128, NTOK / 128), 256, 0, stream>>>(Hs, sw2, out);

  combine_k<<<NTOK, 256, 0, stream>>>(RO, pos, out);
}
